// MultiHeadAttention_76630806495837
// MI455X (gfx1250) — compile-verified, matrix operand order checked by reading, not run on hardware
//
#include <hip/hip_runtime.h>
#include <hip/hip_bf16.h>

#define D_MODEL 1024
#define NHEAD   16
#define DH      64
#define BATCH   4
#define SEQ     2048
#define MTOT    (BATCH * SEQ)      // 8192
#define ATTN_SCALE 0.125f          // 1/sqrt(64)

typedef __bf16 bf16;
typedef __attribute__((ext_vector_type(16))) __bf16 v16bf;
typedef __attribute__((ext_vector_type(8)))  __bf16 v8bf;
typedef __attribute__((ext_vector_type(4)))  __bf16 v4bf;
typedef __attribute__((ext_vector_type(8)))  float  v8f;

typedef __attribute__((ext_vector_type(4))) unsigned int u32x4;
typedef __attribute__((ext_vector_type(8))) int          i32x8;
typedef __attribute__((ext_vector_type(4))) int          i32x4;

// Force a global (AS1) 16-byte load through a loop-carried generic pointer.
typedef __attribute__((address_space(1))) const v8bf* gv8bf_cp;
#define AS1_LOAD_V8(p) (*(gv8bf_cp)(unsigned long long)(p))

#if defined(__has_builtin)
#if __has_builtin(__builtin_amdgcn_tensor_load_to_lds) && \
    __has_builtin(__builtin_amdgcn_s_wait_tensorcnt)
#define HAVE_TDM 1
#endif
#endif
#ifndef HAVE_TDM
#define HAVE_TDM 0
#endif

// Load a 32-byte (16 x bf16) WMMA fragment from LDS as two 16B loads.
__device__ __forceinline__ v16bf ld_frag(const bf16* p) {
    const v8bf* q = (const v8bf*)p;
    v8bf a = q[0];
    v8bf b = q[1];
    v16bf r;
#pragma unroll
    for (int i = 0; i < 8; ++i) { r[i] = a[i]; r[8 + i] = b[i]; }
    return r;
}

__device__ __forceinline__ v8f wmma_bf16(v16bf a, v16bf b, v8f c) {
    return __builtin_amdgcn_wmma_f32_16x16x32_bf16(false, a, false, b,
                                                   (short)0, c, false, false);
}

#if HAVE_TDM
// Issue one TDM 2-D tile load: global (row-major, rowStride elems of 2B)
// -> LDS at byte offset ldsOff, tile tile_d0 x tile_d1 elements, with LDS
// row padding (pad_interval/pad_amount codes per ISA 8.4).
// clang-23 signature: (u32x4 g0, i32x8 g1, i32x4 g2, i32x4 g3, i32x8, i32 cpol)
__device__ __forceinline__ void tdm_load_2d(unsigned ldsOff, const void* gaddr,
                                            unsigned tensor_d0, unsigned tensor_d1,
                                            unsigned tile_d0, unsigned tile_d1,
                                            unsigned rowStride,
                                            unsigned padIntCode, unsigned padAmtCode) {
    unsigned long long ga = (unsigned long long)(size_t)gaddr;
    u32x4 g0;
    g0[0] = 1u;                                           // count=1, user mode
    g0[1] = ldsOff;                                       // lds_addr (bytes)
    g0[2] = (unsigned)(ga & 0xffffffffu);                 // global_addr[31:0]
    g0[3] = (unsigned)((ga >> 32) & 0x01ffffffu)          // global_addr[56:32]
            | (2u << 30);                                 // type = 2 (image)
    i32x8 g1;
    g1[0] = (int)((1u << 16)                              // data_size = 2 bytes
                  | (1u << 20)                            // pad_enable
                  | (padIntCode << 22)                    // pad_interval
                  | (padAmtCode << 25));                  // pad_amount
    g1[1] = (int)((tensor_d0 & 0xffffu) << 16);           // tensor_dim0[15:0]
    g1[2] = (int)(((tensor_d0 >> 16) & 0xffffu)           // tensor_dim0[31:16]
                  | ((tensor_d1 & 0xffffu) << 16));       // tensor_dim1[15:0]
    g1[3] = (int)(((tensor_d1 >> 16) & 0xffffu)           // tensor_dim1[31:16]
                  | ((tile_d0 & 0xffffu) << 16));         // tile_dim0
    g1[4] = (int)(tile_d1 & 0xffffu);                     // tile_dim1 (tile_dim2=0)
    g1[5] = (int)rowStride;                               // tensor_dim0_stride[31:0]
    g1[6] = 0;
    g1[7] = 0;
    i32x4 z4 = {0, 0, 0, 0};
    i32x8 z8 = {0, 0, 0, 0, 0, 0, 0, 0};
    __builtin_amdgcn_tensor_load_to_lds(g0, g1, z4, z4, z8, 0);
}
#endif

// ---------------------------------------------------------------------------
// f32 -> bf16 conversion (vectorized x4), used for x
// ---------------------------------------------------------------------------
__global__ void cvt_f32_to_bf16_v4(const float* __restrict__ in,
                                   bf16* __restrict__ out, int n4) {
    int i = blockIdx.x * blockDim.x + threadIdx.x;
    if (i < n4) {
        float4 f = ((const float4*)in)[i];
        v4bf o;
        o[0] = (bf16)f.x; o[1] = (bf16)f.y; o[2] = (bf16)f.z; o[3] = (bf16)f.w;
        ((v4bf*)out)[i] = o;
    }
}

// ---------------------------------------------------------------------------
// Transpose + convert: W f32 [R][C]  ->  Wt bf16 [C][R].  32x32 LDS tiles.
// ---------------------------------------------------------------------------
__global__ __launch_bounds__(256)
void cvt_transpose_f32_to_bf16(const float* __restrict__ in,
                               bf16* __restrict__ outT, int R, int C) {
    __shared__ bf16 t[32][36];
    const int r0 = blockIdx.y * 32;
    const int c0 = blockIdx.x * 32;
    const int tr = threadIdx.x >> 3;
    const int tc = (threadIdx.x & 7) * 4;

    float4 f = *(const float4*)(in + (size_t)(r0 + tr) * C + c0 + tc);
    t[tc + 0][tr] = (bf16)f.x;
    t[tc + 1][tr] = (bf16)f.y;
    t[tc + 2][tr] = (bf16)f.z;
    t[tc + 3][tr] = (bf16)f.w;
    __syncthreads();
    v4bf o = *(const v4bf*)&t[tr][tc];
    *(v4bf*)(outT + (size_t)(c0 + tr) * R + r0 + tc) = o;
}

// ---------------------------------------------------------------------------
// Per-head transpose of V: vb [B*S, D] -> vbT [B*H*DH, S]   (bf16 -> bf16)
// ---------------------------------------------------------------------------
__global__ __launch_bounds__(256)
void transpose_v_heads(const bf16* __restrict__ vb, bf16* __restrict__ vbT) {
    __shared__ bf16 t[32][36];
    const int bh = blockIdx.z;
    const int b  = bh >> 4;
    const int h  = bh & 15;
    const int s0 = blockIdx.x * 32;
    const int d0 = blockIdx.y * 32;
    const int tr = threadIdx.x >> 3;
    const int tc = (threadIdx.x & 7) * 4;

    v4bf v = *(const v4bf*)(vb + (size_t)(b * SEQ + s0 + tr) * D_MODEL +
                            h * DH + d0 + tc);
#pragma unroll
    for (int j = 0; j < 4; ++j) t[tc + j][tr] = v[j];
    __syncthreads();
    v4bf o = *(const v4bf*)&t[tr][tc];
    *(v4bf*)(vbT + (size_t)(bh * DH + d0 + tr) * SEQ + s0 + tc) = o;
}

// ---------------------------------------------------------------------------
// Tiled WMMA GEMM: C[M,N] = A[M,K] (bf16) @ Wt[N,K]^T (bf16) + bias (f32)
// Block tile 128x128, 256 threads = 8 waves (2x4), wave 64x32.
//
// TDM version: BK=64, double-buffered LDS (74 KB); wave 0 issues
// TENSOR_LOAD_TO_LDS for tile k+1 (A and B descriptors) while all waves
// run 16 WMMAs on tile k. One barrier per K-step.
// Fallback (no TDM builtin): register-staged pipeline, BK=64.
// ---------------------------------------------------------------------------
template <bool OUT_F32>
__global__ __launch_bounds__(256)
void gemm_bf16_wmma(const bf16* __restrict__ A, const bf16* __restrict__ Wt,
                    const float* __restrict__ bias, void* Cout,
                    int M, int N, int K) {
    const int tid  = threadIdx.x;
    const int lane = tid & 31;
    const int wave = tid >> 5;
    const int ln   = lane & 15;
    const int hi   = lane >> 4;

    const int rowBase = blockIdx.y * 128;
    const int colBase = blockIdx.x * 128;
    const int mS = (wave >> 2) * 64;   // 2 wave rows
    const int nS = (wave & 3) * 32;    // 4 wave cols

    v8f acc[4][2];
#pragma unroll
    for (int mi = 0; mi < 4; ++mi)
#pragma unroll
        for (int ni = 0; ni < 2; ++ni)
#pragma unroll
            for (int i = 0; i < 8; ++i) acc[mi][ni][i] = 0.0f;

#if HAVE_TDM
    // Row: 64 bf16 = 128B = 32 DWORDs -> pad_interval code 4; pad 8 bf16 =
    // 4 DWORDs -> pad_amount code 3; LDS row stride = 72 bf16 (16B aligned).
    __shared__ bf16 As[2][128][72];
    __shared__ bf16 Bs[2][128][72];
    const unsigned ldsA = (unsigned)(size_t)&As[0][0][0];
    const unsigned ldsB = (unsigned)(size_t)&Bs[0][0][0];
    const unsigned bufBytes = 128u * 72u * 2u;

    const int NT = K >> 6;  // 64-wide K-steps
    if (tid < 32) {
        tdm_load_2d(ldsA, A + (size_t)rowBase * K, K, M, 64, 128, K, 4, 3);
        tdm_load_2d(ldsB, Wt + (size_t)colBase * K, K, N, 64, 128, K, 4, 3);
    }
    for (int kt = 0; kt < NT; ++kt) {
        const int buf = kt & 1;
        if (tid < 32) __builtin_amdgcn_s_wait_tensorcnt(0);
        __syncthreads();  // tile kt visible; all reads of other buffer done
        if (tid < 32 && kt + 1 < NT) {
            const int nb = (kt + 1) & 1;
            const int k0 = (kt + 1) << 6;
            tdm_load_2d(ldsA + nb * bufBytes, A + (size_t)rowBase * K + k0,
                        K, M, 64, 128, K, 4, 3);
            tdm_load_2d(ldsB + nb * bufBytes, Wt + (size_t)colBase * K + k0,
                        K, N, 64, 128, K, 4, 3);
        }
#pragma unroll
        for (int ks = 0; ks < 2; ++ks) {
            v16bf af[4], bfrag[2];
#pragma unroll
            for (int mi = 0; mi < 4; ++mi)
                af[mi] = ld_frag(&As[buf][mS + mi * 16 + ln][ks * 32 + hi * 16]);
#pragma unroll
            for (int ni = 0; ni < 2; ++ni)
                bfrag[ni] = ld_frag(&Bs[buf][nS + ni * 16 + ln][ks * 32 + hi * 16]);
#pragma unroll
            for (int mi = 0; mi < 4; ++mi)
#pragma unroll
                for (int ni = 0; ni < 2; ++ni)
                    acc[mi][ni] = wmma_bf16(af[mi], bfrag[ni], acc[mi][ni]);
        }
    }
#else
    __shared__ bf16 As[128][72];
    __shared__ bf16 Bs[128][72];
    const bf16* pA[4];
    const bf16* pB[4];
    bf16* sA[4];
    bf16* sB[4];
#pragma unroll
    for (int c = 0; c < 4; ++c) {
        int chunk = tid + c * 256;
        int row = chunk >> 3;
        int c8  = (chunk & 7) * 8;
        pA[c] = A  + (size_t)(rowBase + row) * K + c8;
        pB[c] = Wt + (size_t)(colBase + row) * K + c8;
        sA[c] = &As[row][c8];
        sB[c] = &Bs[row][c8];
    }
    v8bf ra[4], rb[4];
#pragma unroll
    for (int c = 0; c < 4; ++c) {
        ra[c] = AS1_LOAD_V8(pA[c]);
        rb[c] = AS1_LOAD_V8(pB[c]);
    }
    const int NT = K >> 6;
    for (int kt = 0; kt < NT; ++kt) {
#pragma unroll
        for (int c = 0; c < 4; ++c) {
            *(v8bf*)sA[c] = ra[c];
            *(v8bf*)sB[c] = rb[c];
        }
        __syncthreads();
        if (kt + 1 < NT) {
#pragma unroll
            for (int c = 0; c < 4; ++c) {
                pA[c] += 64;
                pB[c] += 64;
                ra[c] = AS1_LOAD_V8(pA[c]);
                rb[c] = AS1_LOAD_V8(pB[c]);
            }
        }
#pragma unroll
        for (int ks = 0; ks < 2; ++ks) {
            v16bf af[4], bfrag[2];
#pragma unroll
            for (int mi = 0; mi < 4; ++mi)
                af[mi] = ld_frag(&As[mS + mi * 16 + ln][ks * 32 + hi * 16]);
#pragma unroll
            for (int ni = 0; ni < 2; ++ni)
                bfrag[ni] = ld_frag(&Bs[nS + ni * 16 + ln][ks * 32 + hi * 16]);
#pragma unroll
            for (int mi = 0; mi < 4; ++mi)
#pragma unroll
                for (int ni = 0; ni < 2; ++ni)
                    acc[mi][ni] = wmma_bf16(af[mi], bfrag[ni], acc[mi][ni]);
        }
        __syncthreads();
    }
#endif

    // Epilogue: C/D layout -> element i of lane: row = i + 8*hi, col = ln
#pragma unroll
    for (int mi = 0; mi < 4; ++mi)
#pragma unroll
        for (int ni = 0; ni < 2; ++ni) {
            int col = colBase + nS + ni * 16 + ln;
            float bv = bias[col];
#pragma unroll
            for (int i = 0; i < 8; ++i) {
                int row = rowBase + mS + mi * 16 + i + 8 * hi;
                float val = acc[mi][ni][i] + bv;
                if (OUT_F32)
                    ((float*)Cout)[(size_t)row * N + col] = val;
                else
                    ((bf16*)Cout)[(size_t)row * N + col] = (bf16)val;
            }
        }
}

// ---------------------------------------------------------------------------
// Fused flash attention per (batch, head): 128-query tile / workgroup,
// 64-row K/V tiles; QK^T and PV on WMMA, online softmax in LDS.
// 256 threads = 8 waves, one 16-row query strip each (full 64 cols).
// K/V tiles register-staged with explicit AS(1) global loads.
// ---------------------------------------------------------------------------
__global__ __launch_bounds__(256)
void attn_fused_wmma(const bf16* __restrict__ Q, const bf16* __restrict__ Kg,
                     const bf16* __restrict__ VTg, bf16* __restrict__ Ctx) {
    __shared__ bf16 Qs[128][72];
    __shared__ bf16 Ks[64][72];
    __shared__ bf16 Vt[64][72];
    __shared__ float Sf[128][68];
    __shared__ bf16 Pb[128][72];
    __shared__ float mrow[128], lrow[128], arow[128];

    const int tid  = threadIdx.x;
    const int lane = tid & 31;
    const int wave = tid >> 5;
    const int ln   = lane & 15;
    const int hi   = lane >> 4;

    const int bh = blockIdx.y;
    const int b  = bh >> 4;
    const int h  = bh & 15;
    const int q0 = blockIdx.x * 128;

    const size_t baseRow = (size_t)b * SEQ;
    const int dOff = h * DH;
    const size_t vtBase = (size_t)bh * DH * SEQ;

    const int mS = wave * 16;   // 8 query strips of 16 rows

    // K/V chunk slots: 64x64 tile = 512 v8bf chunks, 2 per thread
    const int r0c = (tid) >> 3;
    const int r1c = (tid + 256) >> 3;
    const int c80 = (tid & 7) * 8;
    const bf16* pK[2];
    const bf16* pV[2];
    bf16* sK[2];
    bf16* sV[2];
    pK[0] = Kg + (baseRow + r0c) * D_MODEL + dOff + c80;
    pK[1] = Kg + (baseRow + r1c) * D_MODEL + dOff + c80;
    pV[0] = VTg + vtBase + (size_t)r0c * SEQ + c80;
    pV[1] = VTg + vtBase + (size_t)r1c * SEQ + c80;
    sK[0] = &Ks[r0c][c80];  sK[1] = &Ks[r1c][c80];
    sV[0] = &Vt[r0c][c80];  sV[1] = &Vt[r1c][c80];

    // Load Q tile (128x64): 1024 v8bf chunks, 4 per thread
#pragma unroll
    for (int c = 0; c < 4; ++c) {
        int chunk = tid + c * 256;
        int r  = chunk >> 3;
        int c8 = (chunk & 7) * 8;
        *(v8bf*)&Qs[r][c8] =
            AS1_LOAD_V8(Q + (baseRow + q0 + r) * D_MODEL + dOff + c8);
    }
    if (tid < 128) { mrow[tid] = -1e30f; lrow[tid] = 0.0f; }

    v8f o[4];
#pragma unroll
    for (int ni = 0; ni < 4; ++ni)
#pragma unroll
        for (int i = 0; i < 8; ++i) o[ni][i] = 0.0f;

    v8bf rk[2], rv[2];
#pragma unroll
    for (int c = 0; c < 2; ++c) {
        rk[c] = AS1_LOAD_V8(pK[c]);
        rv[c] = AS1_LOAD_V8(pV[c]);
    }
    __syncthreads();

    for (int kv = 0; kv < SEQ; kv += 64) {
#pragma unroll
        for (int c = 0; c < 2; ++c) {
            *(v8bf*)sK[c] = rk[c];
            *(v8bf*)sV[c] = rv[c];
        }
        __syncthreads();

        if (kv + 64 < SEQ) {
#pragma unroll
            for (int c = 0; c < 2; ++c) {
                pK[c] += (size_t)64 * D_MODEL;
                pV[c] += 64;
                rk[c] = AS1_LOAD_V8(pK[c]);
                rv[c] = AS1_LOAD_V8(pV[c]);
            }
        }

        // S = Q K^T : wave strip 16 (q) x 64 (k) = 4 blocks x 2 K-steps
        v8f sacc[4];
#pragma unroll
        for (int ni = 0; ni < 4; ++ni)
#pragma unroll
            for (int i = 0; i < 8; ++i) sacc[ni][i] = 0.0f;
#pragma unroll
        for (int ks = 0; ks < 2; ++ks) {
            v16bf aq = ld_frag(&Qs[mS + ln][ks * 32 + hi * 16]);
#pragma unroll
            for (int ni = 0; ni < 4; ++ni) {
                v16bf bk = ld_frag(&Ks[ni * 16 + ln][ks * 32 + hi * 16]);
                sacc[ni] = wmma_bf16(aq, bk, sacc[ni]);
            }
        }
#pragma unroll
        for (int ni = 0; ni < 4; ++ni)
#pragma unroll
            for (int i = 0; i < 8; ++i)
                Sf[mS + i + 8 * hi][ni * 16 + ln] = sacc[ni][i];
        __syncthreads();

        // Online softmax row update (one thread per query row, 128 rows)
        if (tid < 128) {
            int r = tid;
            float mOld = mrow[r];
            float mNew = mOld;
#pragma unroll 8
            for (int c = 0; c < 64; ++c)
                mNew = fmaxf(mNew, Sf[r][c] * ATTN_SCALE);
            float al = __expf(mOld - mNew);
            float ls = 0.0f;
#pragma unroll 8
            for (int c = 0; c < 64; ++c) {
                float p = __expf(Sf[r][c] * ATTN_SCALE - mNew);
                Pb[r][c] = (bf16)p;
                ls += p;
            }
            mrow[r] = mNew;
            lrow[r] = lrow[r] * al + ls;
            arow[r] = al;
        }
        __syncthreads();

        // Rescale O, then O += P @ V  (strip 16 (q) x 64 (d))
#pragma unroll
        for (int ni = 0; ni < 4; ++ni)
#pragma unroll
            for (int i = 0; i < 8; ++i)
                o[ni][i] *= arow[mS + i + 8 * hi];
#pragma unroll
        for (int ks = 0; ks < 2; ++ks) {
            v16bf ap = ld_frag(&Pb[mS + ln][ks * 32 + hi * 16]);
#pragma unroll
            for (int ni = 0; ni < 4; ++ni) {
                v16bf bv = ld_frag(&Vt[ni * 16 + ln][ks * 32 + hi * 16]);
                o[ni] = wmma_bf16(ap, bv, o[ni]);
            }
        }
        __syncthreads();
    }

    // Normalize by running sum and write context [B*S, D] (bf16)
#pragma unroll
    for (int ni = 0; ni < 4; ++ni) {
        int d = dOff + ni * 16 + ln;
#pragma unroll
        for (int i = 0; i < 8; ++i) {
            int r = mS + i + 8 * hi;
            float inv = 1.0f / lrow[r];
            Ctx[(baseRow + q0 + r) * D_MODEL + d] = (bf16)(o[ni][i] * inv);
        }
    }
}

// ---------------------------------------------------------------------------
extern "C" void kernel_launch(void* const* d_in, const int* in_sizes, int n_in,
                              void* d_out, int out_size, void* d_ws, size_t ws_size,
                              hipStream_t stream) {
    const float* x  = (const float*)d_in[0];
    const float* Wq = (const float*)d_in[1];
    const float* bq = (const float*)d_in[2];
    const float* Wk = (const float*)d_in[3];
    const float* bk = (const float*)d_in[4];
    const float* Wv = (const float*)d_in[5];
    const float* bv = (const float*)d_in[6];
    const float* Wo = (const float*)d_in[7];
    const float* bo = (const float*)d_in[8];
    float* out = (float*)d_out;

    char* ws = (char*)d_ws;
    size_t off = 0;
    bf16* xb  = (bf16*)(ws + off); off += (size_t)MTOT * D_MODEL * 2;
    bf16* wqT = (bf16*)(ws + off); off += (size_t)D_MODEL * D_MODEL * 2;
    bf16* wkT = (bf16*)(ws + off); off += (size_t)D_MODEL * D_MODEL * 2;
    bf16* wvT = (bf16*)(ws + off); off += (size_t)D_MODEL * D_MODEL * 2;
    bf16* woT = (bf16*)(ws + off); off += (size_t)D_MODEL * D_MODEL * 2;
    bf16* qb  = (bf16*)(ws + off); off += (size_t)MTOT * D_MODEL * 2;
    bf16* kb  = (bf16*)(ws + off); off += (size_t)MTOT * D_MODEL * 2;
    bf16* vb  = (bf16*)(ws + off); off += (size_t)MTOT * D_MODEL * 2;
    bf16* vbT = (bf16*)(ws + off); off += (size_t)MTOT * D_MODEL * 2;
    bf16* ctx = (bf16*)(ws + off); off += (size_t)MTOT * D_MODEL * 2;

    {
        int n4 = MTOT * D_MODEL / 4;
        cvt_f32_to_bf16_v4<<<(n4 + 255) / 256, 256, 0, stream>>>(x, xb, n4);
    }
    {
        dim3 tg(D_MODEL / 32, D_MODEL / 32);
        cvt_transpose_f32_to_bf16<<<tg, 256, 0, stream>>>(Wq, wqT, D_MODEL, D_MODEL);
        cvt_transpose_f32_to_bf16<<<tg, 256, 0, stream>>>(Wk, wkT, D_MODEL, D_MODEL);
        cvt_transpose_f32_to_bf16<<<tg, 256, 0, stream>>>(Wv, wvT, D_MODEL, D_MODEL);
        cvt_transpose_f32_to_bf16<<<tg, 256, 0, stream>>>(Wo, woT, D_MODEL, D_MODEL);
    }

    dim3 ggrid(D_MODEL / 128, MTOT / 128);
    gemm_bf16_wmma<false><<<ggrid, 256, 0, stream>>>(xb, wqT, bq, qb,
                                                     MTOT, D_MODEL, D_MODEL);
    gemm_bf16_wmma<false><<<ggrid, 256, 0, stream>>>(xb, wkT, bk, kb,
                                                     MTOT, D_MODEL, D_MODEL);
    gemm_bf16_wmma<false><<<ggrid, 256, 0, stream>>>(xb, wvT, bv, vb,
                                                     MTOT, D_MODEL, D_MODEL);

    {
        dim3 vt(SEQ / 32, DH / 32, BATCH * NHEAD);
        transpose_v_heads<<<vt, 256, 0, stream>>>(vb, vbT);
    }

    dim3 agrid(SEQ / 128, BATCH * NHEAD);   // (16, 64)
    attn_fused_wmma<<<agrid, 256, 0, stream>>>(qb, kb, vbT, ctx);

    gemm_bf16_wmma<true><<<ggrid, 256, 0, stream>>>(ctx, woT, bo, (void*)out,
                                                    MTOT, D_MODEL, D_MODEL);
}
